// GraphEncoder_18236431139069
// MI455X (gfx1250) — matrix operand order repaired and checked by reading, not compiled, not run on hardware
//
#include <hip/hip_runtime.h>

// ---------------------------------------------------------------------------
// CDNA5 / gfx1250 wave32 WMMA implementation of the graph encoder.
// All heavy GEMMs use v_wmma_f32_16x16x32_bf16 (bf16 inputs, f32 accum).
// A-tiles are staged once per block into LDS (ds_load_b128 fragment reads)
// instead of 16x-redundant global loads per wave.
// ---------------------------------------------------------------------------

typedef __attribute__((ext_vector_type(16))) __bf16 bf16x16;
typedef __attribute__((ext_vector_type(8)))  __bf16 bf16x8;
typedef __attribute__((ext_vector_type(8)))  float  floatx8;

union BFrag { bf16x16 v; bf16x8 h[2]; };

constexpr int BB = 4;
constexpr int NN = 10000;
constexpr int EE = 80000;
constexpr int DD = 256;
constexpr int BN = BB * NN;   // 40000 node rows
constexpr int BE = BB * EE;   // 320000 edge rows

__device__ __forceinline__ floatx8 wmma_bf16(bf16x16 a, bf16x16 b, floatx8 c) {
  return __builtin_amdgcn_wmma_f32_16x16x32_bf16(false, a, false, b, (short)0, c,
                                                 false, false);
}

// A fragment (16x32 bf16, M x K): lanes 0-15 -> K base 0, lanes 16-31 -> K base 8;
// vector elems 0..7 = K kb..kb+7, elems 8..15 = K kb+16..kb+23.
__device__ __forceinline__ bf16x16 load_a_frag(const __bf16* arow, int k0, int hlf) {
  int kb = k0 + (hlf ? 8 : 0);
  BFrag f;
  f.h[0] = *(const bf16x8*)(arow + kb);
  f.h[1] = *(const bf16x8*)(arow + kb + 16);
  return f.v;
}

// B fragment (32x16 bf16, K x N) from weights pre-transposed to [N][K]:
// lanes 0-15 -> K base 0, lanes 16-31 -> K base 8; elems 0..7 = K kb..kb+7, elems 8..15 = K kb+16..kb+23, column n = lane&15.
__device__ __forceinline__ bf16x16 load_b_frag(const __bf16* __restrict__ wtrow,
                                               int k0, int hlf) {
  int kb = k0 + (hlf ? 8 : 0);
  BFrag f;
  f.h[0] = *(const bf16x8*)(wtrow + kb);
  f.h[1] = *(const bf16x8*)(wtrow + kb + 16);
  return f.v;
}

__device__ __forceinline__ float block_sum256(float v, float* red) {
  int t = threadIdx.x;
  red[t] = v;
  __syncthreads();
  for (int st = 128; st > 0; st >>= 1) {
    if (t < st) red[t] += red[t + st];
    __syncthreads();
  }
  float r = red[0];
  __syncthreads();
  return r;
}

// ---- weight transpose + f32->bf16 (out [N][Kpad], zero padded K) ------------
__global__ void k_wt(const float* __restrict__ W, __bf16* __restrict__ WT,
                     int K, int N, int Kpad) {
  int i = blockIdx.x * 256 + threadIdx.x;
  if (i >= N * Kpad) return;
  int n = i / Kpad, k = i % Kpad;
  WT[i] = (__bf16)(k < K ? W[(size_t)k * N + n] : 0.0f);
}

// ---- node input projection: h = concat(node,global) @ W_node_in + b ---------
__global__ void k_node_in(const float* __restrict__ nodef,
                          const float* __restrict__ globf,
                          const __bf16* __restrict__ WT,
                          const float* __restrict__ bias,
                          float* __restrict__ h) {
  const int K = 96;
  __shared__ __bf16 atile[16 * 96];
  int mtile = blockIdx.x;            // 2500 tiles over BN
  int t = threadIdx.x;
  // stage concat(node,global) tile -> bf16 LDS once per block
  for (int idx = t; idx < 16 * 96; idx += 512) {
    int row = idx / 96, k = idx - row * 96;
    int m = mtile * 16 + row;
    int b = m / NN;
    float v = (k < 64) ? nodef[(size_t)m * 64 + k]
                       : globf[(size_t)b * 32 + (k - 64)];
    atile[idx] = (__bf16)v;
  }
  __syncthreads();

  int wave = t >> 5, lane = t & 31;  // 16 waves -> 16 ntiles (full 256 cols)
  int hlf = lane >> 4, l16 = lane & 15;
  int n = wave * 16 + l16;
  const __bf16* arow = atile + l16 * K;
  const __bf16* wtrow = WT + (size_t)n * K;
  floatx8 c = {};
#pragma unroll
  for (int k0 = 0; k0 < K; k0 += 32)
    c = wmma_bf16(load_a_frag(arow, k0, hlf), load_b_frag(wtrow, k0, hlf), c);
  float bn = bias[n];
#pragma unroll
  for (int r = 0; r < 8; ++r) {
    int row = mtile * 16 + r + (hlf ? 8 : 0);
    h[(size_t)row * DD + n] = c[r] + bn;
  }
}

// ---- edge projection: e = relu(LN(ef @ W_edge_in + b)) -> bf16 --------------
__global__ void k_edge_proj(const float* __restrict__ ef,
                            const __bf16* __restrict__ WT,
                            const float* __restrict__ bias,
                            const float* __restrict__ lns,
                            const float* __restrict__ lnb,
                            __bf16* __restrict__ ebf) {
  __shared__ float tile[16 * 256];
  __shared__ float mu[16], rsg[16];
  int mtile = blockIdx.x;            // BE/16 tiles
  int wave = threadIdx.x >> 5, lane = threadIdx.x & 31;
  int hlf = lane >> 4, l16 = lane & 15;
  int m = mtile * 16 + l16;
  int n = wave * 16 + l16;
  BFrag fa;
  {
    const float* p = ef + (size_t)m * 16 + (hlf ? 8 : 0);   // K 0..15 valid
#pragma unroll
    for (int i = 0; i < 8; ++i) fa.h[0][i] = (__bf16)p[i];
#pragma unroll
    for (int i = 0; i < 8; ++i) fa.h[1][i] = (__bf16)0.0f;  // K 16..31 zero-pad
  }
  bf16x16 fb = load_b_frag(WT + (size_t)n * 32, 0, hlf);    // WT padded to K=32
  floatx8 c = {};
  c = wmma_bf16(fa.v, fb, c);
  float bn = bias[n];
#pragma unroll
  for (int r = 0; r < 8; ++r)
    tile[(r + (hlf ? 8 : 0)) * 256 + n] = c[r] + bn;
  __syncthreads();
  if (threadIdx.x < 16) {
    float s = 0.f, ss = 0.f;
    for (int j = 0; j < 256; ++j) {
      float v = tile[threadIdx.x * 256 + j];
      s += v; ss += v * v;
    }
    float mean = s * (1.f / 256.f);
    mu[threadIdx.x] = mean;
    rsg[threadIdx.x] = rsqrtf(ss * (1.f / 256.f) - mean * mean + 1e-6f);
  }
  __syncthreads();
  for (int j = threadIdx.x; j < 4096; j += 512) {
    int r = j >> 8, col = j & 255;
    float v = (tile[j] - mu[r]) * rsg[r] * lns[col] + lnb[col];
    v = v > 0.f ? v : 0.f;
    ebf[((size_t)mtile * 16 + r) * DD + col] = (__bf16)v;
  }
}

// ---- degree: deg[b,n] = sum over masked edges with receiver n ---------------
__global__ void k_deg(const int* __restrict__ elist,
                      const unsigned char* __restrict__ emask,
                      float* __restrict__ deg) {
  int i = blockIdx.x * 256 + threadIdx.x;
  if (i >= BE) return;
  if (emask[i]) {
    int r = elist[(size_t)i * 2 + 1];
    int b = i / EE;
    atomicAdd(&deg[b * NN + r], 1.0f);
  }
}

// ---- per-row LN (+optional relu) -> bf16 ------------------------------------
__global__ void k_ln_to_bf16(const float* __restrict__ x,
                             const float* __restrict__ s,
                             const float* __restrict__ bsh,
                             __bf16* __restrict__ out, int do_relu) {
  __shared__ float red[256];
  size_t row = blockIdx.x;
  int t = threadIdx.x;
  float v = x[row * DD + t];
  float mean = block_sum256(v, red) * (1.f / 256.f);
  float d = v - mean;
  float var = block_sum256(d * d, red) * (1.f / 256.f);
  float y = d * rsqrtf(var + 1e-6f) * s[t] + bsh[t];
  if (do_relu) y = y > 0.f ? y : 0.f;
  out[row * DD + t] = (__bf16)y;
}

// ---- GCN message GEMM + masked scatter-add ----------------------------------
// msg = concat(hn[sender], e[edge]) @ W + b, scatter into agg[receiver].
// A-tile (16 rows x 512 bf16 = 16KB) staged once per block in LDS.
__global__ void k_msg(const __bf16* __restrict__ hn,
                      const __bf16* __restrict__ ebf,
                      const __bf16* __restrict__ WT,
                      const float* __restrict__ bias,
                      const int* __restrict__ elist,
                      const unsigned char* __restrict__ emask,
                      float* __restrict__ agg) {
  constexpr int K = 512;
  __shared__ __bf16 atile[16 * K];
  int mtile = blockIdx.x;            // BE/16 = 20000 tiles
  int t = threadIdx.x;
  {
    int row = t >> 5;                // 0..15: tile row
    int seg = t & 31;                // 0..31: 16-byte segment within 256 elems
    int m = mtile * 16 + row;
    int b = m / EE;
    int snd = elist[(size_t)m * 2 + 0];
    if (!emask[m]) snd = 0;          // safe gather (result masked away later)
    const bf16x8* src_h = (const bf16x8*)(hn + ((size_t)b * NN + snd) * DD);
    const bf16x8* src_e = (const bf16x8*)(ebf + (size_t)m * DD);
    bf16x8* dst = (bf16x8*)(atile + row * K);
    dst[seg] = src_h[seg];           // sender features -> K 0..255
    dst[32 + seg] = src_e[seg];      // edge features   -> K 256..511
    // prefetch next tile's edge-feature row (global_prefetch_b8)
    if (mtile + 1 < (int)gridDim.x)
      __builtin_prefetch(ebf + ((size_t)(mtile + 1) * 16 + row) * DD + seg * 8, 0, 1);
  }
  __syncthreads();

  int wave = t >> 5, lane = t & 31;
  int hlf = lane >> 4, l16 = lane & 15;
  int n = wave * 16 + l16;
  const __bf16* arow = atile + l16 * K;
  const __bf16* wtrow = WT + (size_t)n * K;
  floatx8 c = {};
#pragma unroll
  for (int k0 = 0; k0 < K; k0 += 32)
    c = wmma_bf16(load_a_frag(arow, k0, hlf), load_b_frag(wtrow, k0, hlf), c);
  float bn = bias[n];
#pragma unroll
  for (int r = 0; r < 8; ++r) {
    int row = mtile * 16 + r + (hlf ? 8 : 0);
    if (emask[row]) {
      int rcv = elist[(size_t)row * 2 + 1];
      int bb = row / EE;
      atomicAdd(&agg[((size_t)bb * NN + rcv) * DD + n], c[r] + bn);
    }
  }
}

// ---- residual update: h = hn + agg / max(deg,1) -----------------------------
__global__ void k_update(float* __restrict__ h, const __bf16* __restrict__ hn,
                         const float* __restrict__ agg,
                         const float* __restrict__ deg) {
  size_t i = (size_t)blockIdx.x * 256 + threadIdx.x; // one row per block
  float d = deg[blockIdx.x];
  d = d < 1.f ? 1.f : d;
  h[i] = (float)hn[i] + agg[i] * (1.f / d);
}

// ---- bf16 GEMM, K=256, N=256 output (K/V projections) -----------------------
__global__ void k_gemm256(const __bf16* __restrict__ A,
                          const __bf16* __restrict__ WT,
                          const float* __restrict__ bias,
                          float* __restrict__ C) {
  constexpr int K = 256;
  __shared__ __bf16 atile[16 * K];
  int mtile = blockIdx.x;
  int t = threadIdx.x;
  {
    int row = t >> 5, seg = t & 31;  // 32 segs x 8 elems = 256 per row
    const bf16x8* src = (const bf16x8*)(A + ((size_t)mtile * 16 + row) * K);
    ((bf16x8*)(atile + row * K))[seg] = src[seg];
  }
  __syncthreads();

  int wave = t >> 5, lane = t & 31;
  int hlf = lane >> 4, l16 = lane & 15;
  int n = wave * 16 + l16;
  const __bf16* arow = atile + l16 * K;
  const __bf16* wtrow = WT + (size_t)n * K;
  floatx8 c = {};
#pragma unroll
  for (int k0 = 0; k0 < K; k0 += 32)
    c = wmma_bf16(load_a_frag(arow, k0, hlf), load_b_frag(wtrow, k0, hlf), c);
  float bn = bias[n];
#pragma unroll
  for (int r = 0; r < 8; ++r)
    C[((size_t)mtile * 16 + r + (hlf ? 8 : 0)) * DD + n] = c[r] + bn;
}

// ---- agent query: q_in gather, LN, Q = q @ Wq + bq --------------------------
__global__ void k_q(const float* __restrict__ h, const int* __restrict__ agent,
                    const float* __restrict__ lnqs, const float* __restrict__ lnqb,
                    const float* __restrict__ Wq, const float* __restrict__ bq,
                    float* __restrict__ Qb, float* __restrict__ qin) {
  __shared__ float red[256];
  __shared__ float qn[256];
  int b = blockIdx.x, t = threadIdx.x;
  int node = agent[b];
  float v = h[((size_t)b * NN + node) * DD + t];
  qin[b * DD + t] = v;
  float mean = block_sum256(v, red) * (1.f / 256.f);
  float d = v - mean;
  float var = block_sum256(d * d, red) * (1.f / 256.f);
  qn[t] = d * rsqrtf(var + 1e-6f) * lnqs[t] + lnqb[t];
  __syncthreads();
  float acc = bq[t];
  for (int k = 0; k < 256; ++k) acc += qn[k] * Wq[k * 256 + t];
  Qb[b * DD + t] = acc;
}

// ---- masked multi-head attention, online softmax, 1 query -------------------
__global__ void k_attn(const float* __restrict__ Q, const float* __restrict__ Km,
                       const float* __restrict__ Vm,
                       const unsigned char* __restrict__ nmask,
                       float* __restrict__ o) {
  __shared__ float red[256];
  __shared__ float racc[256 * 32];
  int b = blockIdx.x >> 3;
  int hd = blockIdx.x & 7;
  int t = threadIdx.x;
  float qreg[32];
#pragma unroll
  for (int c = 0; c < 32; ++c) qreg[c] = Q[b * DD + hd * 32 + c];
  float mloc = -3.4e38f, sloc = 0.f;
  float acc[32];
#pragma unroll
  for (int c = 0; c < 32; ++c) acc[c] = 0.f;
  const float scale = 0.1767766952966369f;  // 1/sqrt(32)
  for (int n = t; n < NN; n += 256) {
    const float* krow = Km + ((size_t)b * NN + n) * DD + hd * 32;
    float sc = 0.f;
#pragma unroll
    for (int c = 0; c < 32; ++c) sc += qreg[c] * krow[c];
    sc *= scale;
    if (!nmask[b * NN + n]) sc = -1e9f;
    if (sc > mloc) {
      float f = __expf(mloc - sc);
      sloc *= f;
#pragma unroll
      for (int c = 0; c < 32; ++c) acc[c] *= f;
      mloc = sc;
    }
    float p = __expf(sc - mloc);
    sloc += p;
    const float* vrow = Vm + ((size_t)b * NN + n) * DD + hd * 32;
#pragma unroll
    for (int c = 0; c < 32; ++c) acc[c] += p * vrow[c];
  }
  red[t] = mloc; __syncthreads();
  for (int st = 128; st > 0; st >>= 1) { if (t < st) red[t] = fmaxf(red[t], red[t + st]); __syncthreads(); }
  float mg = red[0]; __syncthreads();
  float f = __expf(mloc - mg);
  red[t] = sloc * f; __syncthreads();
  for (int st = 128; st > 0; st >>= 1) { if (t < st) red[t] += red[t + st]; __syncthreads(); }
  float sg = red[0]; __syncthreads();
#pragma unroll
  for (int c = 0; c < 32; ++c) racc[t * 32 + c] = acc[c] * f;
  __syncthreads();
  for (int st = 128; st > 0; st >>= 1) {
    if (t < st) {
#pragma unroll
      for (int c = 0; c < 32; ++c) racc[t * 32 + c] += racc[(t + st) * 32 + c];
    }
    __syncthreads();
  }
  if (t < 32) o[b * DD + hd * 32 + t] = racc[t] / sg;
}

// ---- final: x = relu(LN(q_in + o @ Wo + bo)) --------------------------------
__global__ void k_final(const float* __restrict__ qin, const float* __restrict__ ob,
                        const float* __restrict__ Wo, const float* __restrict__ bo,
                        const float* __restrict__ lnfs, const float* __restrict__ lnfb,
                        float* __restrict__ out) {
  __shared__ float red[256];
  __shared__ float ov[256];
  int b = blockIdx.x, t = threadIdx.x;
  ov[t] = ob[b * DD + t];
  __syncthreads();
  float acc = bo[t];
  for (int k = 0; k < 256; ++k) acc += ov[k] * Wo[k * 256 + t];
  float x = qin[b * DD + t] + acc;
  float mean = block_sum256(x, red) * (1.f / 256.f);
  float d = x - mean;
  float var = block_sum256(d * d, red) * (1.f / 256.f);
  float y = d * rsqrtf(var + 1e-6f) * lnfs[t] + lnfb[t];
  out[b * DD + t] = y > 0.f ? y : 0.f;
}

// ---------------------------------------------------------------------------
extern "C" void kernel_launch(void* const* d_in, const int* in_sizes, int n_in,
                              void* d_out, int out_size, void* d_ws, size_t ws_size,
                              hipStream_t stream) {
  const float* nodef = (const float*)d_in[0];
  const float* edgef = (const float*)d_in[1];
  const float* globf = (const float*)d_in[2];
  const int*   elist = (const int*)d_in[3];
  const unsigned char* emask = (const unsigned char*)d_in[4];
  const unsigned char* nmask = (const unsigned char*)d_in[5];
  const int*   agent = (const int*)d_in[6];
  const float* Wni  = (const float*)d_in[7];
  const float* bni  = (const float*)d_in[8];
  const float* Wei  = (const float*)d_in[9];
  const float* bei  = (const float*)d_in[10];
  const float* lnes = (const float*)d_in[11];
  const float* lneb = (const float*)d_in[12];
  const float* gcnW = (const float*)d_in[13];
  const float* gcnb = (const float*)d_in[14];
  const float* lnsS = (const float*)d_in[15];
  const float* lnsB = (const float*)d_in[16];
  const float* lnqs = (const float*)d_in[17];
  const float* lnqb = (const float*)d_in[18];
  const float* lnks = (const float*)d_in[19];
  const float* lnkb = (const float*)d_in[20];
  const float* Wq   = (const float*)d_in[21];
  const float* bq   = (const float*)d_in[22];
  const float* Wk   = (const float*)d_in[23];
  const float* bk   = (const float*)d_in[24];
  const float* Wv   = (const float*)d_in[25];
  const float* bv   = (const float*)d_in[26];
  const float* Wo   = (const float*)d_in[27];
  const float* bo   = (const float*)d_in[28];
  const float* lnfs = (const float*)d_in[29];
  const float* lnfb = (const float*)d_in[30];
  float* out = (float*)d_out;

  char* ws = (char*)d_ws;
  size_t off = 0;
  auto alloc = [&](size_t bytes) -> void* {
    void* p = ws + off;
    off = (off + bytes + 255) & ~(size_t)255;
    return p;
  };

  __bf16* WnodeT = (__bf16*)alloc((size_t)256 * 96 * 2);
  __bf16* WedgeT = (__bf16*)alloc((size_t)256 * 32 * 2);
  __bf16* WgcnT  = (__bf16*)alloc((size_t)3 * 256 * 512 * 2);
  __bf16* WkT    = (__bf16*)alloc((size_t)256 * 256 * 2);
  __bf16* WvT    = (__bf16*)alloc((size_t)256 * 256 * 2);
  float*  h      = (float*)alloc((size_t)BN * DD * 4);
  __bf16* ebf    = (__bf16*)alloc((size_t)BE * DD * 2);
  __bf16* hn     = (__bf16*)alloc((size_t)BN * DD * 2);
  float*  agg    = (float*)alloc((size_t)BN * DD * 4);
  float*  deg    = (float*)alloc((size_t)BN * 4);
  float*  Km     = (float*)alloc((size_t)BN * DD * 4);
  float*  Vm     = (float*)alloc((size_t)BN * DD * 4);
  __bf16* kln    = (__bf16*)alloc((size_t)BN * DD * 2);
  float*  Qb     = (float*)alloc((size_t)BB * DD * 4);
  float*  qin    = (float*)alloc((size_t)BB * DD * 4);
  float*  ob     = (float*)alloc((size_t)BB * DD * 4);

  // weights -> transposed bf16
  k_wt<<<96, 256, 0, stream>>>(Wni, WnodeT, 96, 256, 96);
  k_wt<<<32, 256, 0, stream>>>(Wei, WedgeT, 16, 256, 32);
  for (int i = 0; i < 3; ++i)
    k_wt<<<512, 256, 0, stream>>>(gcnW + (size_t)i * 512 * 256,
                                  WgcnT + (size_t)i * 256 * 512, 512, 256, 512);
  k_wt<<<256, 256, 0, stream>>>(Wk, WkT, 256, 256, 256);
  k_wt<<<256, 256, 0, stream>>>(Wv, WvT, 256, 256, 256);

  k_node_in<<<BN / 16, 512, 0, stream>>>(nodef, globf, WnodeT, bni, h);
  k_edge_proj<<<BE / 16, 512, 0, stream>>>(edgef, WedgeT, bei, lnes, lneb, ebf);

  hipMemsetAsync(deg, 0, (size_t)BN * 4, stream);
  k_deg<<<(BE + 255) / 256, 256, 0, stream>>>(elist, emask, deg);

  for (int i = 0; i < 3; ++i) {
    k_ln_to_bf16<<<BN, 256, 0, stream>>>(h, lnsS + i * 256, lnsB + i * 256, hn, 1);
    hipMemsetAsync(agg, 0, (size_t)BN * DD * 4, stream);
    k_msg<<<BE / 16, 512, 0, stream>>>(hn, ebf, WgcnT + (size_t)i * 256 * 512,
                                       gcnb + i * 256, elist, emask, agg);
    k_update<<<BN, 256, 0, stream>>>(h, hn, agg, deg);
  }

  k_ln_to_bf16<<<BN, 256, 0, stream>>>(h, lnks, lnkb, kln, 0);
  k_gemm256<<<BN / 16, 512, 0, stream>>>(kln, WkT, bk, Km);
  k_gemm256<<<BN / 16, 512, 0, stream>>>(kln, WvT, bv, Vm);

  k_q<<<BB, 256, 0, stream>>>(h, agent, lnqs, lnqb, Wq, bq, Qb, qin);
  k_attn<<<BB * 8, 256, 0, stream>>>(Qb, Km, Vm, nmask, ob);
  k_final<<<BB, 256, 0, stream>>>(qin, ob, Wo, bo, lnfs, lnfb, out);

  (void)in_sizes; (void)n_in; (void)out_size; (void)ws_size;
}
